// RobotEncoder_32057635897382
// MI455X (gfx1250) — matrix operand order repaired and checked by reading, not compiled, not run on hardware
//
#include <hip/hip_runtime.h>
#include <hip/hip_bf16.h>
#include <math.h>

// ---------------- problem constants ----------------
#define BGRAPH 64
#define VPG    2048
#define NNODE  4096      // B*N
#define NEDGE  8064      // B*2*E, E=63 -> 126 directed edges per graph

typedef __attribute__((ext_vector_type(16))) __bf16 bf16x16;
typedef __attribute__((ext_vector_type(8)))  __bf16 bf16x8;
typedef __attribute__((ext_vector_type(8)))  float  v8f;
typedef __attribute__((ext_vector_type(4)))  unsigned v4u;
typedef __attribute__((ext_vector_type(4)))  int      v4i;
typedef __attribute__((ext_vector_type(8)))  int      v8i;

#define USE_TDM __has_builtin(__builtin_amdgcn_tensor_load_to_lds)

__device__ __forceinline__ unsigned short f2bf(float f) {
  unsigned u = __float_as_uint(f);
  u += 0x7FFFu + ((u >> 16) & 1u);          // round-to-nearest-even
  return (unsigned short)(u >> 16);
}
__device__ __forceinline__ unsigned encf(float f) {
  unsigned u = __float_as_uint(f);
  return (u & 0x80000000u) ? ~u : (u | 0x80000000u);
}
__device__ __forceinline__ float decf(unsigned u) {
  unsigned b = (u & 0x80000000u) ? (u & 0x7FFFFFFFu) : ~u;
  return __uint_as_float(b);
}
__device__ __forceinline__ bf16x16 frag16(const unsigned short* p0, const unsigned short* p1) {
  bf16x8 lo = *(const bf16x8*)p0;
  bf16x8 hi = *(const bf16x8*)p1;
  return __builtin_shufflevector(lo, hi, 0,1,2,3,4,5,6,7,8,9,10,11,12,13,14,15);
}

__device__ __forceinline__ void wait_tensor0() {
#if __has_builtin(__builtin_amdgcn_s_wait_tensorcnt)
  __builtin_amdgcn_s_wait_tensorcnt(0);
#else
  asm volatile("s_wait_tensorcnt 0x0" ::: "memory");
#endif
}

#if USE_TDM
// Issue one TDM descriptor: 3D tile (dim0=64 ch contiguous, dim1=16 rows @1152,
// dim2=4 planes @20736, bf16) global -> LDS.  Per CDNA5 ISA 8.3-8.6.
__device__ __forceinline__ void tdm_load_tile(unsigned lds_addr, unsigned long gaddr)
{
  v4u g0; v8i g1; v4i g2 = {4, 0, 0, 0}; v4i g3 = {0, 0, 0, 0};
  g0[0] = 1u;                                   // count=1, user descriptor
  g0[1] = lds_addr;                             // lds_addr (bytes)
  g0[2] = (unsigned)(gaddr & 0xFFFFFFFFu);      // global_addr lo
  g0[3] = (unsigned)((gaddr >> 32) & 0x1FFFFFFu) | (2u << 30);  // addr hi | type=2
  g1[0] = (int)(1u << 16);                      // data_size=1 (2 bytes/elem)
  g1[1] = (int)(64u << 16);                     // tensor_dim0 = 64 (lo16)
  g1[2] = (int)(16u << 16);                     // tensor_dim1 = 16 (lo16)
  g1[3] = (int)(64u << 16);                     // tile_dim0 = 64
  g1[4] = (int)(16u | (4u << 16));              // tile_dim1=16, tile_dim2=4
  g1[5] = (int)1152;                            // tensor_dim0_stride (18*64)
  g1[6] = (int)((20736u & 0xFFFFu) << 16);      // tensor_dim1_stride lo16 (18*18*64)
  g1[7] = 0;                                    // tensor_dim1_stride hi
#if defined(__clang_major__) && __clang_major__ >= 23
  v8i z8 = {0,0,0,0,0,0,0,0};
  __builtin_amdgcn_tensor_load_to_lds(g0, g1, g2, g3, z8, 0);
#else
  __builtin_amdgcn_tensor_load_to_lds(g0, g1, g2, g3, 0);
#endif
}
#endif

// =====================================================================
// Generic WMMA GEMM: C[M,N](ldc) = A[M,K] @ W[K,N] + bias
// Tile 64(M) x NT(N) per block; each wave owns one 16-col strip and 4
// M-subtiles -> 4 v_wmma per K-chunk off a single B fragment.
// =====================================================================
template<int NT, int INACT>
__global__ __launch_bounds__(2*NT)
void gemm_bf16_wmma(const float* __restrict__ A, const float* __restrict__ W,
                    const float* __restrict__ bias, float* __restrict__ C,
                    int M, int K, int N, int ldc)
{
  __shared__ __align__(16) unsigned short sA[64][32];
  __shared__ __align__(16) unsigned short sB[NT][32];
  const int tid  = threadIdx.x;
  const int wave = tid >> 5;
  const int lane = tid & 31;
  const int m0 = blockIdx.x * 64;
  const int n0 = blockIdx.y * NT;
  const int nthr = 2 * NT;
  v8f acc[4] = {};

  for (int kc = 0; kc < K; kc += 32) {
    for (int i = tid; i < 64 * 32; i += nthr) {
      int r = i >> 5, kk = i & 31, k = kc + kk;
      float v = (k < K) ? A[(size_t)(m0 + r) * K + k] : 0.f;
      if (INACT == 2) v = v > 0.f ? v : 0.2f * v;   // leaky_relu(0.2) on input
      sA[r][kk] = f2bf(v);
    }
    for (int i = tid; i < NT * 32; i += nthr) {
      int n = i >> 5, kk = i & 31, k = kc + kk;
      float v = (k < K) ? W[(size_t)k * N + (n0 + n)] : 0.f;
      sB[n][kk] = f2bf(v);
    }
    if (kc + 32 < K) __builtin_prefetch(&A[(size_t)m0 * K + kc + 32], 0, 0);
    __syncthreads();

    const int l15 = lane & 15;
    const int g0 = (lane < 16) ? 0 : 8;             // A swizzle (ISA 7.12.2)
    const int kb = (lane < 16) ? 0 : 8;             // B swizzle
    bf16x16 b = frag16(&sB[wave * 16 + l15][kb], &sB[wave * 16 + l15][kb + 16]);
#pragma unroll
    for (int ms = 0; ms < 4; ++ms) {
      bf16x16 a = frag16(&sA[ms * 16 + l15][g0], &sA[ms * 16 + l15][g0 + 16]);
      acc[ms] = __builtin_amdgcn_wmma_f32_16x16x32_bf16(false, a, false, b,
                                                        (short)0, acc[ms], false, false);
    }
    __syncthreads();
  }

  const int n = n0 + wave * 16 + (lane & 15);
  const float bs = bias ? bias[n] : 0.f;
  const int rb = (lane < 16) ? 0 : 8;               // C layout: VGPR r -> M=r / r+8
#pragma unroll
  for (int ms = 0; ms < 4; ++ms)
#pragma unroll
    for (int r = 0; r < 8; ++r)
      C[(size_t)(m0 + ms * 16 + rb + r) * ldc + n] = acc[ms][r] + bs;
}

// =====================================================================
// 3D conv 3x3x3, C=64->64: 27 shifted WMMA GEMM taps over the padded
// bf16 grid [B][18][18][18][64].  Block = (b, z, 4 x-planes) -> 64 pos x
// 64 ch.  A tile per tap loaded by the Tensor Data Mover (wave 0 issues,
// s_wait_tensorcnt 0, then workgroup barrier).
// =====================================================================
__global__ __launch_bounds__(256)
void conv3d_wmma(const unsigned short* __restrict__ pad,
                 const float* __restrict__ Kw, const float* __restrict__ bias,
                 float* __restrict__ out)
{
  __shared__ __align__(16) unsigned short sA[64][64];
  __shared__ __align__(16) unsigned short sB[64][64];
  const int tid = threadIdx.x, wave = tid >> 5, lane = tid & 31;
  const int bid = blockIdx.x;                  // b*64 + z*4 + xtile
  const int xtile = bid & 3, z = (bid >> 2) & 15, b = bid >> 6;
  const int xbase = xtile * 4;
  const int mb = (wave >> 2) * 2;              // this wave's 2 M-subtiles
  const int nb = (wave & 3) * 16;              // this wave's 16-col strip
  v8f acc[2] = {};

  for (int tap = 0; tap < 27; ++tap) {
    const int dx = tap / 9, dy = (tap / 3) % 3, dz = tap % 3;
    // weights for the whole tap, transposed [n][k]
    for (int i = tid; i < 64 * 64; i += 256) {
      int n = i >> 6, k = i & 63;
      sB[n][k] = f2bf(Kw[((size_t)tap * 64 + k) * 64 + n]);
    }
    // A tile: 4 x-planes x 16 y-rows x 64 ch from padded grid
    const size_t tile0 = ((((size_t)b * 18 + (xbase + dx)) * 18 + dy) * 18 + (z + dz)) * 64;
#if USE_TDM
    if (wave == 0) {
      tdm_load_tile((unsigned)(size_t)&sA[0][0],
                    (unsigned long)(size_t)(pad + tile0));
      wait_tensor0();
    }
#else
    for (int i = tid; i < 64 * 64; i += 256) {
      int r = i >> 6, c = i & 63;                  // r = (x_local*16 + y)
      sA[r][c] = pad[tile0 + (size_t)(r >> 4) * 20736 + (size_t)(r & 15) * 1152 + c];
    }
#endif
    __syncthreads();
#pragma unroll
    for (int kc = 0; kc < 64; kc += 32) {
      const int l15 = lane & 15;
      const int g0 = (lane < 16) ? 0 : 8;
      const int kb = (lane < 16) ? 0 : 8;
      bf16x16 bb = frag16(&sB[nb + l15][kc + kb], &sB[nb + l15][kc + kb + 16]);
#pragma unroll
      for (int ms = 0; ms < 2; ++ms) {
        bf16x16 a = frag16(&sA[(mb + ms) * 16 + l15][kc + g0],
                           &sA[(mb + ms) * 16 + l15][kc + g0 + 16]);
        acc[ms] = __builtin_amdgcn_wmma_f32_16x16x32_bf16(false, a, false, bb,
                                                          (short)0, acc[ms], false, false);
      }
    }
    __syncthreads();
  }
  const int n = nb + (lane & 15);
  const float bs = bias[n];
  const int rb = (lane < 16) ? 0 : 8;
#pragma unroll
  for (int ms = 0; ms < 2; ++ms)
#pragma unroll
    for (int r = 0; r < 8; ++r) {
      int p = (mb + ms) * 16 + rb + r;
      int x = xbase + (p >> 4), y = p & 15;
      float v = acc[ms][r] + bs;
      out[((((size_t)b * 16 + x) * 16 + y) * 16 + z) * 64 + n] = v > 0.f ? v : 0.f;
    }
}

// ---------------- voxel encoder: MLP + scatter-max ----------------
__global__ void voxel_encode_scatter(const float* __restrict__ com,
                                     const float* __restrict__ vpos,
                                     const float* __restrict__ vfeat,
                                     const float* __restrict__ W,
                                     const float* __restrict__ bvec,
                                     unsigned* __restrict__ grid)
{
  int i = blockIdx.x * blockDim.x + threadIdx.x;
  if (i >= BGRAPH * VPG) return;
  int b = i >> 11;
  float f0 = vfeat[i*4+0], f1 = vfeat[i*4+1], f2 = vfeat[i*4+2], f3 = vfeat[i*4+3];
  int id[3];
#pragma unroll
  for (int d = 0; d < 3; ++d) {
    float nv = (vpos[i*3+d] - com[b*3+d]) * 0.5f + 0.5f;
    nv = fminf(fmaxf(nv, 0.f), 1.f - 1e-6f);
    int ix = (int)floorf(nv * 16.f);
    id[d] = ix > 15 ? 15 : (ix < 0 ? 0 : ix);
  }
  size_t base = ((size_t)b * 4096 + ((id[0]*16 + id[1])*16 + id[2])) * 64;
#pragma unroll 4
  for (int c = 0; c < 64; ++c) {
    float h = f0*W[c] + f1*W[64+c] + f2*W[128+c] + f3*W[192+c] + bvec[c];
    h = fmaxf(h, 0.f);                       // >=0, so raw bits are uint-monotone
    atomicMax(&grid[base + c], __float_as_uint(h));
  }
}

// ---------------- fp32 grid -> zero-padded bf16 grid ----------------
__global__ void pad_grid_bf16(const float* __restrict__ g, unsigned short* __restrict__ pad)
{
  size_t i = (size_t)blockIdx.x * blockDim.x + threadIdx.x;
  const size_t total = (size_t)BGRAPH * 18 * 18 * 18 * 64;
  if (i >= total) return;
  int c = i & 63; size_t t = i >> 6;
  int pz = t % 18; t /= 18; int py = t % 18; t /= 18; int px = t % 18; int b = (int)(t / 18);
  float v = 0.f;
  if (px >= 1 && px <= 16 && py >= 1 && py <= 16 && pz >= 1 && pz <= 16)
    v = g[((((size_t)b * 16 + (px-1)) * 16 + (py-1)) * 16 + (pz-1)) * 64 + c];
  pad[i] = f2bf(v);
}

// ---------------- trilinear sample ----------------
__global__ void trilinear_sample_k(const float* __restrict__ com,
                                   const float* __restrict__ npos,
                                   const float* __restrict__ g,
                                   float* __restrict__ samp)
{
  int i = blockIdx.x * blockDim.x + threadIdx.x;
  if (i >= NNODE * 64) return;
  int c = i & 63; int t = i >> 6; int n = t & 63; int b = t >> 6;
  float fr[3]; int i0[3];
#pragma unroll
  for (int d = 0; d < 3; ++d) {
    float nv = (npos[(size_t)(b*64+n)*3 + d] - com[b*3+d]) * 0.5f + 0.5f;
    nv = fminf(fmaxf(nv, 0.f), 1.f);
    float cc = nv * 15.f;
    int ii = (int)floorf(cc);
    ii = ii > 14 ? 14 : (ii < 0 ? 0 : ii);
    i0[d] = ii; fr[d] = cc - (float)ii;
  }
  float acc = 0.f;
#pragma unroll
  for (int dx = 0; dx < 2; ++dx)
#pragma unroll
    for (int dy = 0; dy < 2; ++dy)
#pragma unroll
      for (int dz = 0; dz < 2; ++dz) {
        float w = (dx ? fr[0] : 1.f-fr[0]) * (dy ? fr[1] : 1.f-fr[1]) * (dz ? fr[2] : 1.f-fr[2]);
        acc += w * g[((((size_t)b*16 + i0[0]+dx)*16 + i0[1]+dy)*16 + i0[2]+dz)*64 + c];
      }
  samp[(size_t)(b*64+n)*64 + c] = acc;
}

// ---------------- concat node features into x[:,64:78] ----------------
__global__ void concat_nodefeat(const float* __restrict__ nf, float* __restrict__ x)
{
  int i = blockIdx.x * blockDim.x + threadIdx.x;
  if (i >= NNODE * 14) return;
  int c = i % 14, n = i / 14;
  x[(size_t)n * 78 + 64 + c] = nf[i];
}

// ---------------- per-edge attention logits (one wave per edge) ----------------
__global__ void edge_logits(const int* __restrict__ edges, const float* __restrict__ Q,
                            const float* __restrict__ Kt, const float* __restrict__ E,
                            float* __restrict__ logit, int dout, float scale)
{
  int e = blockIdx.x * (blockDim.x >> 5) + (threadIdx.x >> 5);
  if (e >= NEDGE) return;
  int lane = threadIdx.x & 31;
  int src = edges[e], dst = edges[NEDGE + e];
  float s = 0.f;
  for (int c = lane; c < dout; c += 32)
    s += Q[(size_t)dst*dout + c] * (Kt[(size_t)src*dout + c] + E[(size_t)e*dout + c]);
#pragma unroll
  for (int m = 16; m; m >>= 1) s += __shfl_xor(s, m);
  if (lane == 0) logit[e] = s * scale;
}

__global__ void seg_max_k(const int* __restrict__ edges, const float* __restrict__ logit,
                          unsigned* __restrict__ m)
{
  int e = blockIdx.x * blockDim.x + threadIdx.x;
  if (e >= NEDGE) return;
  atomicMax(&m[edges[NEDGE + e]], encf(logit[e]));
}

__global__ void edge_exp_k(const int* __restrict__ edges, const float* __restrict__ logit,
                           const unsigned* __restrict__ m, float* __restrict__ ex,
                           float* __restrict__ den)
{
  int e = blockIdx.x * blockDim.x + threadIdx.x;
  if (e >= NEDGE) return;
  int dst = edges[NEDGE + e];
  unsigned u = m[dst];
  float mm = (u == 0u) ? 0.f : decf(u);
  if (!(mm > -1e37f && mm < 1e37f)) mm = 0.f;
  float v = __expf(logit[e] - mm);
  ex[e] = v;
  atomicAdd(&den[dst], v);
}

__global__ void edge_agg_k(const int* __restrict__ edges, const float* __restrict__ V,
                           const float* __restrict__ E, const float* __restrict__ ex,
                           float* __restrict__ agg, int dout)
{
  int e = blockIdx.x * (blockDim.x >> 5) + (threadIdx.x >> 5);
  if (e >= NEDGE) return;
  int lane = threadIdx.x & 31;
  int src = edges[e], dst = edges[NEDGE + e];
  float w = ex[e];
  for (int c = lane; c < dout; c += 32)
    atomicAdd(&agg[(size_t)dst*dout + c],
              (V[(size_t)src*dout + c] + E[(size_t)e*dout + c]) * w);
}

__global__ void tconv_final(const float* __restrict__ agg, const float* __restrict__ den,
                            const float* __restrict__ S, float* __restrict__ out, int dout)
{
  int i = blockIdx.x * blockDim.x + threadIdx.x;
  if (i >= NNODE * dout) return;
  int node = i / dout;
  out[i] = agg[i] / (den[node] + 1e-16f) + S[i];
}

// ---------------- global max pool over nodes ----------------
__global__ void pool_max(const float* __restrict__ out2, float* __restrict__ pooled)
{
  int i = blockIdx.x * blockDim.x + threadIdx.x;
  if (i >= BGRAPH * 512) return;
  int c = i % 512, b = i / 512;
  float m = -INFINITY;
  for (int n = 0; n < 64; ++n)
    m = fmaxf(m, out2[((size_t)(b*64 + n)) * 512 + c]);
  pooled[i] = m;
}

// ---------------- index-tensor outputs, as floats ----------------
__global__ void tail_write(const int* __restrict__ edges, float* __restrict__ out)
{
  int i = blockIdx.x * blockDim.x + threadIdx.x;
  const int n0 = 4096, n1 = n0 + 8064, n2 = n1 + 16128, n3 = n2 + 4032, n4 = n3 + 8064;
  if (i >= n4) return;
  float v; int o;
  if (i < n0)      { v = (float)(i >> 6);               o = 1081344 + i; }       // node_batch
  else if (i < n1) { int j = i - n0; v = (float)(j/126); o = 1085440 + j; }      // edge_batch
  else if (i < n2) { int j = i - n1; v = (float)edges[j]; o = 1093504 + j; }     // edges
  else if (i < n3) { int j = i - n2; v = (float)(j/63);  o = 1109632 + j; }      // uniq edge batch
  else { int j = i - n3; int r = j / 4032; int t = j % 4032; int b = t / 63; int u = t % 63;
         v = (float)edges[r*8064 + b*126 + u]; o = 1113664 + j; }                // uniq edges
  out[o] = v;
}

// =====================================================================
// host side
// =====================================================================
static void run_gemm(const float* A, const float* W, const float* bias, float* C,
                     int M, int K, int N, int ldc, int lrelu_in, hipStream_t s)
{
  if (N % 128 == 0) {
    dim3 g(M / 64, N / 128);
    if (lrelu_in) gemm_bf16_wmma<128, 2><<<g, 256, 0, s>>>(A, W, bias, C, M, K, N, ldc);
    else          gemm_bf16_wmma<128, 0><<<g, 256, 0, s>>>(A, W, bias, C, M, K, N, ldc);
  } else {
    dim3 g(M / 64, N / 64);
    if (lrelu_in) gemm_bf16_wmma<64, 2><<<g, 128, 0, s>>>(A, W, bias, C, M, K, N, ldc);
    else          gemm_bf16_wmma<64, 0><<<g, 128, 0, s>>>(A, W, bias, C, M, K, N, ldc);
  }
}

extern "C" void kernel_launch(void* const* d_in, const int* in_sizes, int n_in,
                              void* d_out, int out_size, void* d_ws, size_t ws_size,
                              hipStream_t stream)
{
  (void)in_sizes; (void)n_in; (void)out_size; (void)ws_size;
  const float* com   = (const float*)d_in[0];
  const float* vpos  = (const float*)d_in[1];
  const float* vfeat = (const float*)d_in[2];
  const float* npos  = (const float*)d_in[3];
  const float* nfeat = (const float*)d_in[4];
  const int*   edges = (const int*)  d_in[5];
  const float* efeat = (const float*)d_in[6];
  const float* encW  = (const float*)d_in[7];
  const float* encB  = (const float*)d_in[8];
  const float* convK = (const float*)d_in[9];
  const float* convB = (const float*)d_in[10];
  const float* decW  = (const float*)d_in[11];
  const float* decB  = (const float*)d_in[12];
  float* out = (float*)d_out;

  // ---- workspace layout ----
  char* ws = (char*)d_ws;
  size_t off = 0;
  auto take = [&](size_t bytes) -> char* {
    char* p = ws + off; off = (off + bytes + 255) & ~(size_t)255; return p;
  };
  unsigned*       grid_u = (unsigned*)      take((size_t)BGRAPH*4096*64*4);   // fp32 grid / conv out
  unsigned short* padg   = (unsigned short*)take((size_t)BGRAPH*18*18*18*64*2);
  float* samp = (float*)take((size_t)NNODE*64*4);
  float* x0   = (float*)take((size_t)NNODE*78*4);
  float* bq_  = (float*)take((size_t)NNODE*512*4);   // Q
  float* bk_  = (float*)take((size_t)NNODE*512*4);   // K
  float* bv_  = (float*)take((size_t)NNODE*512*4);   // V
  float* bs_  = (float*)take((size_t)NNODE*512*4);   // skip
  float* bE   = (float*)take((size_t)NEDGE*512*4);   // edge transform
  float* blog = (float*)take((size_t)NEDGE*4);
  float* bex  = (float*)take((size_t)NEDGE*4);
  unsigned* bm = (unsigned*)take((size_t)NNODE*4);
  float* bden = (float*)take((size_t)NNODE*4);
  float* bagg = (float*)take((size_t)NNODE*512*4);
  float* outA = (float*)take((size_t)NNODE*512*4);   // layer-1 out
  float* outB = (float*)take((size_t)NNODE*512*4);   // layer-2 out
  float* gridf = (float*)grid_u;

  // 1) voxel encoder: MLP + scatter-max into zeroed grid
  hipMemsetAsync(grid_u, 0, (size_t)BGRAPH*4096*64*4, stream);
  voxel_encode_scatter<<<(BGRAPH*VPG + 255)/256, 256, 0, stream>>>(com, vpos, vfeat, encW, encB, grid_u);

  // 2) pad/convert to bf16, then 3D conv via 27 WMMA GEMM taps (TDM-staged A tiles)
  {
    size_t total = (size_t)BGRAPH*18*18*18*64;
    pad_grid_bf16<<<(unsigned)((total + 255)/256), 256, 0, stream>>>(gridf, padg);
  }
  conv3d_wmma<<<BGRAPH*16*4, 256, 0, stream>>>(padg, convK, convB, gridf);

  // 3) trilinear sample + decoder GEMM (writes into x columns [0,64) of stride 78)
  trilinear_sample_k<<<(NNODE*64 + 255)/256, 256, 0, stream>>>(com, npos, gridf, samp);
  run_gemm(samp, decW, decB, x0, NNODE, 64, 64, 78, 0, stream);
  concat_nodefeat<<<(NNODE*14 + 255)/256, 256, 0, stream>>>(nfeat, x0);

  // 4) three TransformerConv layers
  const int din_l[3]  = {78, 256, 512};
  const int dout_l[3] = {256, 512, 512};
  const float* Ain[3] = {x0, out /*out1*/, outA};
  float* Lout[3] = {out /*out1 -> d_out[0:]*/, outA, outB};
  for (int l = 0; l < 3; ++l) {
    const int base = 13 + l*9;
    const float* Wq = (const float*)d_in[base+0]; const float* bq = (const float*)d_in[base+1];
    const float* Wk = (const float*)d_in[base+2]; const float* bk = (const float*)d_in[base+3];
    const float* Wv = (const float*)d_in[base+4]; const float* bv = (const float*)d_in[base+5];
    const float* We = (const float*)d_in[base+6];
    const float* Wsk= (const float*)d_in[base+7]; const float* bsk= (const float*)d_in[base+8];
    const int din = din_l[l], dout = dout_l[l];
    const int inact = (l > 0) ? 1 : 0;                 // leaky_relu(0.2) between layers

    run_gemm(Ain[l], Wq,  bq,  bq_, NNODE, din, dout, dout, inact, stream);
    run_gemm(Ain[l], Wk,  bk,  bk_, NNODE, din, dout, dout, inact, stream);
    run_gemm(Ain[l], Wv,  bv,  bv_, NNODE, din, dout, dout, inact, stream);
    run_gemm(Ain[l], Wsk, bsk, bs_, NNODE, din, dout, dout, inact, stream);
    run_gemm(efeat,  We,  nullptr, bE, NEDGE, 9, dout, dout, 0, stream);  // edge transform

    float scale = 1.0f / sqrtf((float)dout);
    edge_logits<<<(NEDGE + 7)/8, 256, 0, stream>>>(edges, bq_, bk_, bE, blog, dout, scale);

    hipMemsetAsync(bm,   0, (size_t)NNODE*4, stream);
    hipMemsetAsync(bden, 0, (size_t)NNODE*4, stream);
    hipMemsetAsync(bagg, 0, (size_t)NNODE*dout*4, stream);
    seg_max_k <<<(NEDGE + 255)/256, 256, 0, stream>>>(edges, blog, bm);
    edge_exp_k<<<(NEDGE + 255)/256, 256, 0, stream>>>(edges, blog, bm, bex, bden);
    edge_agg_k<<<(NEDGE + 7)/8, 256, 0, stream>>>(edges, bv_, bE, bex, bagg, dout);
    tconv_final<<<(NNODE*dout + 255)/256, 256, 0, stream>>>(bagg, bden, bs_, Lout[l], dout);
  }

  // 5) global max pool of layer-2 output -> d_out[1048576:]
  pool_max<<<(BGRAPH*512 + 255)/256, 256, 0, stream>>>(outB, out + 1048576);

  // 6) index outputs
  tail_write<<<(4096 + 8064 + 16128 + 4032 + 8064 + 255)/256, 256, 0, stream>>>(edges, out);
}
